// MultiHeadAttention_72224170050032
// MI455X (gfx1250) — matrix operand order repaired and checked by reading, not compiled, not run on hardware
//
#include <hip/hip_runtime.h>

// ---------------------------------------------------------------------------
// MI455X (gfx1250) multi-head attention, bf16 WMMA + async global->LDS copies.
// ---------------------------------------------------------------------------

typedef __attribute__((ext_vector_type(16))) __bf16 bf16x16;
typedef __attribute__((ext_vector_type(8)))  float  f32x8;

#define NB     4
#define QLEN   2048
#define KLEN   2048
#define DMODEL 1024
#define NHEAD  16
#define DK     64

// --- CDNA5 async global->LDS copy (ASYNCcnt-tracked, no VGPR bounce) -------
// LDS destination address = low 32 bits of the generic pointer to __shared__
// (flat->LDS truncation, ISA section 10.2).

__device__ __forceinline__ void async_copy_b128(const void* gptr, void* lptr) {
  unsigned int lds_addr = (unsigned int)(unsigned long long)lptr;
  asm volatile("global_load_async_to_lds_b128 %0, %1, off"
               :
               : "v"(lds_addr), "v"(gptr)
               : "memory");
}

__device__ __forceinline__ void wait_async() {
  asm volatile("s_wait_asynccnt 0" ::: "memory");
}

// --- WMMA fragment helpers (layouts per CDNA5 ISA 7.12.2) -------

// A-matrix 16x32 bf16: lane (m = lane&15, hi = lane>>4), vgpr v holds
// k = (v>>2)*16 + hi*8 + (v&3)*2 and k+1.  rowp points at LDS row of tile.
__device__ __forceinline__ bf16x16 load_frag_A(const __bf16* rowp, int hi, int koff) {
  bf16x16 a;
#pragma unroll
  for (int v = 0; v < 8; ++v) {
    int kb = ((v >> 2) << 4) + (hi << 3) + ((v & 3) << 1) + koff;
    a[2 * v]     = rowp[kb];
    a[2 * v + 1] = rowp[kb + 1];
  }
  return a;
}

// B-matrix 32x16 bf16: lane (n = lane&15, hi), vgpr v holds
// k = (v>>2)*16 + hi*8 + (v&3)*2 and k+1.
// rowp points at an LDS row laid out n-major with k contiguous.
__device__ __forceinline__ bf16x16 load_frag_B(const __bf16* rowp, int hi, int koff) {
  bf16x16 b;
#pragma unroll
  for (int v = 0; v < 8; ++v) {
    int kb = ((v >> 2) << 4) + (hi << 3) + ((v & 3) << 1) + koff;
    b[2 * v]     = rowp[kb];
    b[2 * v + 1] = rowp[kb + 1];
  }
  return b;
}

__device__ __forceinline__ f32x8 wmma_bf16(bf16x16 a, bf16x16 b, f32x8 c) {
  return __builtin_amdgcn_wmma_f32_16x16x32_bf16(
      /*neg_a=*/false, a, /*neg_b=*/false, b,
      /*c_mod=*/(short)0, c, /*reuse_a=*/false, /*reuse_b=*/false);
}

// --- fp32 -> bf16 conversion ----------------------------------------------

__global__ void f32_to_bf16_kernel(const float* __restrict__ src,
                                   __bf16* __restrict__ dst, int n) {
  int i = blockIdx.x * blockDim.x + threadIdx.x;
  int stride = gridDim.x * blockDim.x;
  for (; i < n; i += stride) dst[i] = (__bf16)src[i];
}

// --- Per-head projection GEMM:  Out[n,h,l,e] = sum_d X[n*l, d] * W[h,d,e] --
// grid: (M/64, NHEAD), block 256 (8 waves). 64x64 output tile per block.

__global__ __launch_bounds__(256) void proj_kernel(
    const __bf16* __restrict__ X,   // [M x DMODEL], M = NB*LEN
    const __bf16* __restrict__ W,   // [NHEAD][DMODEL][DK]
    __bf16* __restrict__ Out,       // [NB][NHEAD][LEN][DK]
    int LEN) {
  __shared__ __align__(16) __bf16 Ads[64][40];  // [m][d], d contiguous
  __shared__ __align__(16) __bf16 Bds[64][40];  // [e][d], d contiguous (W^T)

  const int t = threadIdx.x;
  const int h = blockIdx.y;
  const int mBase = blockIdx.x * 64;
  const int wave = t >> 5, lane = t & 31;
  const int m = lane & 15, hi = lane >> 4, n = lane & 15;
  const int wrow = (wave & 3) * 16, wcol = (wave >> 2) * 32;

  const int aRow = t >> 2, aC8 = (t & 3) * 8;   // A-tile loader: 8 bf16 each
  const int bD = t >> 3, bE0 = (t & 7) * 8;     // B-tile loader (transpose)

  f32x8 acc0 = {0.f, 0.f, 0.f, 0.f, 0.f, 0.f, 0.f, 0.f};
  f32x8 acc1 = {0.f, 0.f, 0.f, 0.f, 0.f, 0.f, 0.f, 0.f};

  for (int d0 = 0; d0 < DMODEL; d0 += 32) {
    // A tile: async global->LDS (linear copy, 16B per lane)
    async_copy_b128(&X[(size_t)(mBase + aRow) * DMODEL + d0 + aC8],
                    &Ads[aRow][aC8]);
    // B tile: transpose on the fly (scatter -> regular ds stores)
    {
      uint4 tmp = *(const uint4*)&W[(size_t)h * DMODEL * DK +
                                    (size_t)(d0 + bD) * DK + bE0];
      const __bf16* tv = (const __bf16*)&tmp;
#pragma unroll
      for (int j = 0; j < 8; ++j) Bds[bE0 + j][bD] = tv[j];
    }
    if (d0 + 32 < DMODEL) {
      __builtin_prefetch(&X[(size_t)(mBase + aRow) * DMODEL + d0 + 32 + aC8], 0, 1);
    }
    wait_async();
    __syncthreads();

    bf16x16 a  = load_frag_A(&Ads[wrow + m][0], hi, 0);
    bf16x16 b0 = load_frag_B(&Bds[wcol + n][0], hi, 0);
    bf16x16 b1 = load_frag_B(&Bds[wcol + 16 + n][0], hi, 0);
    acc0 = wmma_bf16(a, b0, acc0);
    acc1 = wmma_bf16(a, b1, acc1);
    __syncthreads();
  }

#pragma unroll
  for (int r = 0; r < 8; ++r) {
    int row = mBase + wrow + hi * 8 + r;
    int nb = row / LEN, l = row % LEN;
    size_t o = (((size_t)nb * NHEAD + h) * LEN + l) * DK;
    Out[o + wcol + n]      = (__bf16)acc0[r];
    Out[o + wcol + 16 + n] = (__bf16)acc1[r];
  }
}

// --- Flash attention per (q-tile, head, batch) -----------------------------
// grid: (QLEN/64, NHEAD, NB), block 256. Mask applied BEFORE 1/sqrt(dk)
// scaling, exactly as in the reference. Online softmax, fp32 accumulators.

__global__ __launch_bounds__(256) void attn_kernel(
    const __bf16* __restrict__ Qi,  // [NB][NHEAD][QLEN][DK]
    const __bf16* __restrict__ Ki,  // [NB][NHEAD][KLEN][DK]
    const __bf16* __restrict__ Vi,  // [NB][NHEAD][KLEN][DK]
    const int*    __restrict__ mask,// [NB][QLEN][KLEN]
    __bf16* __restrict__ concat) {  // [NB][QLEN][NHEAD*DK]
  __shared__ __align__(16) __bf16 Qds[64][72];  // [q][e]
  __shared__ __align__(16) __bf16 Kds[64][72];  // [k][e]
  __shared__ __align__(16) __bf16 Vds[64][72];  // [e][k] (transposed)
  __shared__ __align__(16) __bf16 Pds[64][72];  // [q][k] bf16 probabilities
  __shared__ float Sds[64][66];                 // [q][k] fp32 scores
  __shared__ float mRow[64], lRow[64], aRowS[64];

  const int t = threadIdx.x;
  const int qBase = blockIdx.x * 64;
  const int h = blockIdx.y, nb = blockIdx.z;
  const int wave = t >> 5, lane = t & 31;
  const int m = lane & 15, hi = lane >> 4, n = lane & 15;
  const int wrow = (wave & 3) * 16, wcol = (wave >> 2) * 32;

  const size_t headOff = (((size_t)nb * NHEAD + h) * KLEN) * DK;
  const int ldRow = t >> 2, ldE0 = (t & 3) * 16;

  {  // Q tile resident for the whole block: async global->LDS
    const __bf16* src = &Qi[headOff + (size_t)(qBase + ldRow) * DK + ldE0];
    async_copy_b128(&src[0], &Qds[ldRow][ldE0]);
    async_copy_b128(&src[8], &Qds[ldRow][ldE0 + 8]);
  }
  if (t < 64) { mRow[t] = -3.0e38f; lRow[t] = 0.f; }

  f32x8 accO0 = {0.f, 0.f, 0.f, 0.f, 0.f, 0.f, 0.f, 0.f};
  f32x8 accO1 = {0.f, 0.f, 0.f, 0.f, 0.f, 0.f, 0.f, 0.f};
  wait_async();
  __syncthreads();

  for (int kt = 0; kt < KLEN; kt += 64) {
    {  // K tile ([k][e]) async; V tile transposed ([e][k]) via ds scatter
      const __bf16* ks = &Ki[headOff + (size_t)(kt + ldRow) * DK + ldE0];
      async_copy_b128(&ks[0], &Kds[ldRow][ldE0]);
      async_copy_b128(&ks[8], &Kds[ldRow][ldE0 + 8]);
      uint4 tv0 = *(const uint4*)&Vi[headOff + (size_t)(kt + ldRow) * DK + ldE0];
      uint4 tv1 = *(const uint4*)&Vi[headOff + (size_t)(kt + ldRow) * DK + ldE0 + 8];
      const __bf16* p0 = (const __bf16*)&tv0;
      const __bf16* p1 = (const __bf16*)&tv1;
#pragma unroll
      for (int j = 0; j < 8; ++j) Vds[ldE0 + j][ldRow] = p0[j];
#pragma unroll
      for (int j = 0; j < 8; ++j) Vds[ldE0 + 8 + j][ldRow] = p1[j];
    }
    if (kt + 64 < KLEN) {  // prefetch next tile into L2 while we compute
      __builtin_prefetch(&Ki[headOff + (size_t)(kt + 64 + ldRow) * DK + ldE0], 0, 1);
      __builtin_prefetch(&Vi[headOff + (size_t)(kt + 64 + ldRow) * DK + ldE0], 0, 1);
    }
    wait_async();
    __syncthreads();

    // S = Q K^T on this wave's 16x32 region (K-dim = DK = 64 -> 2 steps)
    f32x8 s0 = {0.f, 0.f, 0.f, 0.f, 0.f, 0.f, 0.f, 0.f};
    f32x8 s1 = {0.f, 0.f, 0.f, 0.f, 0.f, 0.f, 0.f, 0.f};
#pragma unroll
    for (int e0 = 0; e0 < DK; e0 += 32) {
      bf16x16 a  = load_frag_A(&Qds[wrow + m][0], hi, e0);
      bf16x16 b0 = load_frag_B(&Kds[wcol + n][0], hi, e0);
      bf16x16 b1 = load_frag_B(&Kds[wcol + 16 + n][0], hi, e0);
      s0 = wmma_bf16(a, b0, s0);
      s1 = wmma_bf16(a, b1, s1);
    }

    // mask (before scaling, like the reference), scale, stage into LDS
    const float SC = 0.125f;  // 1/sqrt(64)
#pragma unroll
    for (int r = 0; r < 8; ++r) {
      int q = wrow + hi * 8 + r;
      size_t mo = ((size_t)nb * QLEN + (qBase + q)) * KLEN + kt;
      float v0 = (mask[mo + wcol + n]      == 0) ? -1.0e20f : s0[r];
      float v1 = (mask[mo + wcol + 16 + n] == 0) ? -1.0e20f : s1[r];
      Sds[q][wcol + n]      = v0 * SC;
      Sds[q][wcol + 16 + n] = v1 * SC;
    }
    __syncthreads();

    // online softmax: one thread per q row over this 64-wide tile
    if (t < 64) {
      float mOld = mRow[t];
      float mNew = mOld;
#pragma unroll 8
      for (int c = 0; c < 64; ++c) mNew = fmaxf(mNew, Sds[t][c]);
      float al = __expf(mOld - mNew);
      float sum = 0.f;
#pragma unroll 8
      for (int c = 0; c < 64; ++c) {
        float pv = __expf(Sds[t][c] - mNew);
        Pds[t][c] = (__bf16)pv;
        sum += pv;
      }
      mRow[t]  = mNew;
      lRow[t]  = lRow[t] * al + sum;
      aRowS[t] = al;
    }
    __syncthreads();

    // rescale running O, then O += P * V
#pragma unroll
    for (int r = 0; r < 8; ++r) {
      float al = aRowS[wrow + hi * 8 + r];
      accO0[r] *= al;
      accO1[r] *= al;
    }
#pragma unroll
    for (int k0 = 0; k0 < 64; k0 += 32) {
      bf16x16 a  = load_frag_A(&Pds[wrow + m][0], hi, k0);
      bf16x16 b0 = load_frag_B(&Vds[wcol + n][0], hi, k0);
      bf16x16 b1 = load_frag_B(&Vds[wcol + 16 + n][0], hi, k0);
      accO0 = wmma_bf16(a, b0, accO0);
      accO1 = wmma_bf16(a, b1, accO1);
    }
    __syncthreads();
  }

  // normalize and store into the concat layout [n][q][h*DK + e]
#pragma unroll
  for (int r = 0; r < 8; ++r) {
    int q = wrow + hi * 8 + r;
    float inv = 1.0f / lRow[q];
    size_t o = ((size_t)nb * QLEN + (qBase + q)) * (NHEAD * DK) + (size_t)h * DK;
    concat[o + wcol + n]      = (__bf16)(accO0[r] * inv);
    concat[o + wcol + 16 + n] = (__bf16)(accO1[r] * inv);
  }
}

// --- Output projection: out[m][d] = sum_c concat[m][c]*WO_w[d][c] + b[d] ---
// grid: (M/64, DMODEL/64), block 256. WO_w rows are already k-contiguous.

__global__ __launch_bounds__(256) void outproj_kernel(
    const __bf16* __restrict__ A,    // [M][DMODEL] concat
    const __bf16* __restrict__ Wo,   // [DMODEL][DMODEL] (row d, col c)
    const float*  __restrict__ bias, // [DMODEL]
    float* __restrict__ out) {       // [M][DMODEL]
  __shared__ __align__(16) __bf16 Ads[64][40];  // [m][c]
  __shared__ __align__(16) __bf16 Bds[64][40];  // [d][c]

  const int t = threadIdx.x;
  const int mBase = blockIdx.x * 64;
  const int dBase = blockIdx.y * 64;
  const int wave = t >> 5, lane = t & 31;
  const int m = lane & 15, hi = lane >> 4, n = lane & 15;
  const int wrow = (wave & 3) * 16, wcol = (wave >> 2) * 32;
  const int aRow = t >> 2, aC8 = (t & 3) * 8;

  f32x8 acc0 = {0.f, 0.f, 0.f, 0.f, 0.f, 0.f, 0.f, 0.f};
  f32x8 acc1 = {0.f, 0.f, 0.f, 0.f, 0.f, 0.f, 0.f, 0.f};

  for (int c0 = 0; c0 < DMODEL; c0 += 32) {
    async_copy_b128(&A[(size_t)(mBase + aRow) * DMODEL + c0 + aC8],
                    &Ads[aRow][aC8]);
    async_copy_b128(&Wo[(size_t)(dBase + aRow) * DMODEL + c0 + aC8],
                    &Bds[aRow][aC8]);
    wait_async();
    __syncthreads();

    bf16x16 a  = load_frag_A(&Ads[wrow + m][0], hi, 0);
    bf16x16 b0 = load_frag_B(&Bds[wcol + n][0], hi, 0);
    bf16x16 b1 = load_frag_B(&Bds[wcol + 16 + n][0], hi, 0);
    acc0 = wmma_bf16(a, b0, acc0);
    acc1 = wmma_bf16(a, b1, acc1);
    __syncthreads();
  }

#pragma unroll
  for (int r = 0; r < 8; ++r) {
    int row = mBase + wrow + hi * 8 + r;
    int c0 = dBase + wcol + n;
    int c1 = dBase + wcol + 16 + n;
    out[(size_t)row * DMODEL + c0] = acc0[r] + bias[c0];
    out[(size_t)row * DMODEL + c1] = acc1[r] + bias[c1];
  }
}

// ---------------------------------------------------------------------------

extern "C" void kernel_launch(void* const* d_in, const int* in_sizes, int n_in,
                              void* d_out, int out_size, void* d_ws, size_t ws_size,
                              hipStream_t stream) {
  (void)in_sizes; (void)n_in; (void)out_size; (void)ws_size;

  const float* K    = (const float*)d_in[0];
  const float* Q    = (const float*)d_in[1];
  const float* V    = (const float*)d_in[2];
  const int*   mask = (const int*)d_in[3];
  const float* WQ   = (const float*)d_in[4];
  const float* WK   = (const float*)d_in[5];
  const float* WV   = (const float*)d_in[6];
  const float* WOw  = (const float*)d_in[7];
  const float* WOb  = (const float*)d_in[8];
  float* out = (float*)d_out;

  const size_t XSZ = (size_t)NB * QLEN * DMODEL;        // 8.4M elems
  const size_t WSZ = (size_t)NHEAD * DMODEL * DK;       // 1.0M elems
  const size_t HSZ = (size_t)NB * NHEAD * QLEN * DK;    // 8.4M elems
  const size_t OSZ = (size_t)DMODEL * DMODEL;           // 1.0M elems

  char* base = (char*)d_ws;
  size_t off = 0;
  auto take = [&](size_t elems) -> __bf16* {
    __bf16* r = (__bf16*)(base + off);
    off += ((elems * 2 + 255) & ~(size_t)255);
    return r;
  };
  __bf16* Qb  = take(XSZ);
  __bf16* Kb  = take(XSZ);
  __bf16* Vb  = take(XSZ);
  __bf16* WQb = take(WSZ);
  __bf16* WKb = take(WSZ);
  __bf16* WVb = take(WSZ);
  __bf16* Qi  = take(HSZ);
  __bf16* Ki  = take(HSZ);
  __bf16* Vi  = take(HSZ);
  __bf16* Cc  = take(XSZ);
  __bf16* WoB = take(OSZ);

  f32_to_bf16_kernel<<<1024, 256, 0, stream>>>(Q, Qb, (int)XSZ);
  f32_to_bf16_kernel<<<1024, 256, 0, stream>>>(K, Kb, (int)XSZ);
  f32_to_bf16_kernel<<<1024, 256, 0, stream>>>(V, Vb, (int)XSZ);
  f32_to_bf16_kernel<<<256, 256, 0, stream>>>(WQ, WQb, (int)WSZ);
  f32_to_bf16_kernel<<<256, 256, 0, stream>>>(WK, WKb, (int)WSZ);
  f32_to_bf16_kernel<<<256, 256, 0, stream>>>(WV, WVb, (int)WSZ);
  f32_to_bf16_kernel<<<256, 256, 0, stream>>>(WOw, WoB, (int)OSZ);

  dim3 pj((NB * QLEN) / 64, NHEAD);
  proj_kernel<<<pj, 256, 0, stream>>>(Qb, WQb, Qi, QLEN);
  proj_kernel<<<pj, 256, 0, stream>>>(Kb, WKb, Ki, KLEN);
  proj_kernel<<<pj, 256, 0, stream>>>(Vb, WVb, Vi, KLEN);

  attn_kernel<<<dim3(QLEN / 64, NHEAD, NB), 256, 0, stream>>>(Qi, Ki, Vi, mask, Cc);

  outproj_kernel<<<dim3((NB * QLEN) / 64, DMODEL / 64), 256, 0, stream>>>(
      Cc, WoB, WOb, out);
}
